// MaskedGAT_76768245449445
// MI455X (gfx1250) — matrix operand order repaired and checked by reading, not compiled, not run on hardware
//
#include <hip/hip_runtime.h>
#include <hip/hip_bf16.h>

// ---------------------------------------------------------------------------
// Masked multi-head GAT, N=6144, HID=128, HEADS=4 — CDNA5 (gfx1250) WMMA path
// ---------------------------------------------------------------------------

#define N_NODES   6144
#define HID       128
#define HEADS     4
#define QKV_COLS  (HID * HEADS)          // 512
#define MASK_WPR  (N_NODES / 32)         // 192 u32 words per mask row

typedef __bf16 bf16_t;
typedef __attribute__((ext_vector_type(16))) __bf16 v16bf;
typedef __attribute__((ext_vector_type(8)))  __bf16 v8bf;
typedef __attribute__((ext_vector_type(8)))  float  v8f;

// ---- fp32 -> bf16 round-to-nearest-even (weights / activations) --------
__device__ __forceinline__ bf16_t f2bf(float f) {
    unsigned int u = __builtin_bit_cast(unsigned int, f);
    u += 0x7FFFu + ((u >> 16) & 1u);
    unsigned short s = (unsigned short)(u >> 16);
    return __builtin_bit_cast(bf16_t, s);
}
// ---- fast truncating fp32 -> bf16 (probabilities in [0,1]) -------------
__device__ __forceinline__ bf16_t f2bf_rz(float f) {
    unsigned int u = __builtin_bit_cast(unsigned int, f);
    return __builtin_bit_cast(bf16_t, (unsigned short)(u >> 16));
}

// same-wave LDS store->load ordering: LDS is in-order per wave (ISA 11/15.5),
// we only need to stop the compiler from reordering; the dscnt wait is cheap
// insurance.
__device__ __forceinline__ void lds_fence_wave() {
    asm volatile("s_wait_dscnt 0x0" ::: "memory");
}

// ---- WMMA fragment loaders (ISA 7.12.2 layouts, wave32) ----------------
// A (16x32 bf16): lane L -> row = L&15 ; elems 0..7 -> K = 8*(L>>4)+j,
//                 elems 8..15 -> K = 16 + 8*(L>>4) + (j-8)
__device__ __forceinline__ v16bf load_a_frag(const bf16_t* A, int lda,
                                             int row0, int k0, int lane) {
    int r  = lane & 15;
    int kb = (lane >> 4) << 3;            // 0 or 8
    const bf16_t* p = A + (size_t)(row0 + r) * lda + k0 + kb;
    v8bf lo = *(const v8bf*)p;            // 16B
    v8bf hi = *(const v8bf*)(p + 16);     // 16B
    v16bf a;
#pragma unroll
    for (int j = 0; j < 8; ++j) { a[j] = lo[j]; a[j + 8] = hi[j]; }
    return a;
}

// B (32x16 bf16), sourced from Bt[col][k] (column-major over K):
// lane L -> col = L&15 ; elems 0..7 -> K = 8*(L>>4)+j,
//                 elems 8..15 -> K = 16 + 8*(L>>4) + (j-8)
__device__ __forceinline__ v16bf load_b_frag(const bf16_t* Bt, int ldb,
                                             int col0, int k0, int lane) {
    int c  = lane & 15;
    int ko = (lane >> 4) << 3;            // 0 or 8
    const bf16_t* p = Bt + (size_t)(col0 + c) * ldb + k0 + ko;
    v8bf lo = *(const v8bf*)p;            // 16B
    v8bf hi = *(const v8bf*)(p + 16);     // 16B
    v16bf b;
#pragma unroll
    for (int j = 0; j < 8; ++j) { b[j] = lo[j]; b[j + 8] = hi[j]; }
    return b;
}

__device__ __forceinline__ v8f wmma_bf16(v16bf a, v16bf b, v8f c) {
    return __builtin_amdgcn_wmma_f32_16x16x32_bf16(
        false, a, false, b, (short)0, c, false, false);
}

// ---------------------------------------------------------------------------
// 0) utility kernels: mask build + bf16 conversions
// ---------------------------------------------------------------------------
__global__ void zero_u32_kernel(unsigned int* p, int n) {
    int i = blockIdx.x * blockDim.x + threadIdx.x;
    if (i < n) p[i] = 0u;
}

__global__ void edge_scatter_kernel(const long long* ei, int E, unsigned int* mask) {
    int i = blockIdx.x * blockDim.x + threadIdx.x;
    if (i >= E) return;
    int src = (int)ei[i];
    int dst = (int)ei[(size_t)E + i];
    atomicOr(&mask[(size_t)src * MASK_WPR + (dst >> 5)], 1u << (dst & 31));
}

__global__ void conv_f2bf_kernel(const float* in, bf16_t* out, int n) {
    int i = blockIdx.x * blockDim.x + threadIdx.x;
    if (i < n) out[i] = f2bf(in[i]);
}

// W is (K rows x C cols) fp32 row-major; Wt is (C x K) bf16: Wt[c][k] = W[k][c]
__global__ void conv_transpose_kernel(const float* W, bf16_t* Wt, int K, int C) {
    int i = blockIdx.x * blockDim.x + threadIdx.x;
    if (i >= K * C) return;
    int c = i / K, k = i % K;
    Wt[i] = f2bf(W[(size_t)k * C + c]);
}

// ---------------------------------------------------------------------------
// 1) fused QKV projection: [Q|K|V] = Hb(6144x128) @ Wt_all(1536x128)^T
//    8 waves / block, one wave -> one 16x16 tile; V stored transposed
// ---------------------------------------------------------------------------
__global__ __launch_bounds__(256)
void proj_qkv_kernel(const bf16_t* __restrict__ Hb, const bf16_t* __restrict__ Wt,
                     bf16_t* __restrict__ Qb, bf16_t* __restrict__ Kb,
                     bf16_t* __restrict__ Vt) {
    int lane = threadIdx.x & 31;
    int wid  = threadIdx.x >> 5;
    int n0 = blockIdx.x * 16;
    int ct = blockIdx.y * 8 + wid;        // 0..95 -> col tile in 3*512
    int c0 = ct * 16;
    v8f acc = {};
#pragma unroll
    for (int k0 = 0; k0 < HID; k0 += 32) {
        v16bf a = load_a_frag(Hb, HID, n0, k0, lane);
        v16bf b = load_b_frag(Wt, HID, c0, k0, lane);
        acc = wmma_bf16(a, b, acc);
    }
    int half = lane >> 4, cl = lane & 15;
    if (ct < 32) {
        int c = c0 + cl;
#pragma unroll
        for (int r = 0; r < 8; ++r)
            Qb[(size_t)(n0 + r + half * 8) * QKV_COLS + c] = f2bf(acc[r]);
    } else if (ct < 64) {
        int c = c0 - 512 + cl;
#pragma unroll
        for (int r = 0; r < 8; ++r)
            Kb[(size_t)(n0 + r + half * 8) * QKV_COLS + c] = f2bf(acc[r]);
    } else {
        int c = c0 - 1024 + cl;
#pragma unroll
        for (int r = 0; r < 8; ++r)
            Vt[(size_t)c * N_NODES + n0 + r + half * 8] = f2bf(acc[r]);
    }
}

// ---------------------------------------------------------------------------
// 2) masked flash attention. Block = 128 threads = 4 waves = 4 heads,
//    one 16-row query tile per block; key strips of 32, online softmax.
//    Waves are fully decoupled: the LDS P-tile is private to each wave, so
//    only compiler fences (+ same-wave in-order LDS) are needed, no barriers.
// ---------------------------------------------------------------------------
__global__ __launch_bounds__(128)
void attn_kernel(const bf16_t* __restrict__ Qb, const bf16_t* __restrict__ Kb,
                 const bf16_t* __restrict__ Vt, const unsigned int* __restrict__ mask,
                 bf16_t* __restrict__ Attn) {
    __shared__ __align__(32) bf16_t ldsP[HEADS][16 * 32];
    int lane = threadIdx.x & 31;
    int h    = threadIdx.x >> 5;          // head
    int n0   = blockIdx.x * 16;
    int half = lane >> 4, cl = lane & 15;
    bf16_t* P = ldsP[h];
    // 1/sqrt(128) * log2(e): scores go straight into exp2
    const float sscale = 0.08838834764831845f * 1.4426950408889634f;

    v16bf aq[4];
#pragma unroll
    for (int kk = 0; kk < 4; ++kk)
        aq[kk] = load_a_frag(Qb, QKV_COLS, n0, h * HID + kk * 32, lane);

    v16bf ones;
#pragma unroll
    for (int j = 0; j < 16; ++j) ones[j] = (bf16_t)1.0f;

    v8f o[8];
    float rm[8], rl[8];
#pragma unroll
    for (int j = 0; j < 8; ++j) o[j] = (v8f){};
#pragma unroll
    for (int r = 0; r < 8; ++r) { rm[r] = -1e30f; rl[r] = 0.0f; }

    for (int m0 = 0; m0 < N_NODES; m0 += 32) {
        // ---- scores: S[16x32] = Q(16x128) . K^T(128x32), two 16-col tiles
        v8f s0 = {}, s1 = {};
#pragma unroll
        for (int kk = 0; kk < 4; ++kk) {
            v16bf bk0 = load_b_frag(Kb, QKV_COLS, m0,      h * HID + kk * 32, lane);
            v16bf bk1 = load_b_frag(Kb, QKV_COLS, m0 + 16, h * HID + kk * 32, lane);
            s0 = wmma_bf16(aq[kk], bk0, s0);
            s1 = wmma_bf16(aq[kk], bk1, s1);
        }
        // prefetch next key/value strip (speculative; tail runs past Kb/Vt
        // into adjacent workspace regions, still valid memory)
        __builtin_prefetch(Kb + (size_t)(m0 + 32 + cl) * QKV_COLS + h * HID, 0, 3);
        __builtin_prefetch(Vt + (size_t)(h * HID + (lane << 2)) * N_NODES + m0 + 32, 0, 3);

        // lane L holds mask word of row (n0 + L&15) covering cols m0..m0+31
        unsigned int w = mask[(size_t)(n0 + cl) * MASK_WPR + (m0 >> 5)];

        float t0[8], t1[8], mn[8];
        bool up = false;
#pragma unroll
        for (int r = 0; r < 8; ++r) {
            int row = r + (half << 3);                 // C-layout row 0..15
            unsigned int wr = __shfl(w, row, 32);
            bool b0 = (wr >> cl) & 1u;
            bool b1 = (wr >> (cl + 16)) & 1u;
            float v0 = b0 ? s0[r] * sscale : -1e30f;
            float v1 = b1 ? s1[r] * sscale : -1e30f;
            float mt = fmaxf(v0, v1);
            mt = fmaxf(mt, __shfl_xor(mt, 1, 32));
            mt = fmaxf(mt, __shfl_xor(mt, 2, 32));
            mt = fmaxf(mt, __shfl_xor(mt, 4, 32));
            mt = fmaxf(mt, __shfl_xor(mt, 8, 32));
            float m2 = fmaxf(rm[r], mt);
            up |= (m2 > rm[r]);
            t0[r] = b0 ? exp2f(v0 - m2) : 0.0f;
            t1[r] = b1 ? exp2f(v1 - m2) : 0.0f;
            mn[r] = m2;
        }
        if (__ballot(up)) {                            // wave-uniform branch
#pragma unroll
            for (int r = 0; r < 8; ++r) {
                float corr = exp2f(rm[r] - mn[r]);
                rm[r] = mn[r];
                rl[r] *= corr;
#pragma unroll
                for (int j = 0; j < 8; ++j) o[j][r] *= corr;
            }
        }
        // write P tile (C layout -> LDS); previous iteration's reads are
        // older same-wave LDS ops, hardware keeps them in order.
        asm volatile("" ::: "memory");
#pragma unroll
        for (int r = 0; r < 8; ++r) {
            int row = r + (half << 3);
            P[row * 32 + cl]      = f2bf_rz(t0[r]);
            P[row * 32 + cl + 16] = f2bf_rz(t1[r]);
        }
        lds_fence_wave();   // stores ahead of loads; re-read in A layout

        v16bf ap;
        {
            int kb = half << 3;
            const bf16_t* pp = P + cl * 32 + kb;
            v8bf lo = *(const v8bf*)pp;
            v8bf hi = *(const v8bf*)(pp + 16);
#pragma unroll
            for (int j = 0; j < 8; ++j) { ap[j] = lo[j]; ap[j + 8] = hi[j]; }
        }
        // row sums of P in one matrix op: P(16x32) x ones(32x16)
        v8f sums = {};
        sums = wmma_bf16(ap, ones, sums);
#pragma unroll
        for (int r = 0; r < 8; ++r) rl[r] += sums[r];
#pragma unroll
        for (int j = 0; j < 8; ++j) {
            v16bf bv = load_b_frag(Vt, N_NODES, h * HID + j * 16, m0, lane);
            o[j] = wmma_bf16(ap, bv, o[j]);
        }
    }
    // ---- normalize and write Attn[n][h*128 + d] (bf16)
#pragma unroll
    for (int r = 0; r < 8; ++r) {
        int row = n0 + r + (half << 3);
        float inv = 1.0f / rl[r];
#pragma unroll
        for (int j = 0; j < 8; ++j)
            Attn[(size_t)row * QKV_COLS + h * HID + j * 16 + cl] = f2bf(o[j][r] * inv);
    }
}

// ---------------------------------------------------------------------------
// 3) output projection: out = Attn(6144x512) @ Wp(512x128) + bp   (fp32 out)
//    8 waves / block cover all 128 output columns of a 16-row tile
// ---------------------------------------------------------------------------
__global__ __launch_bounds__(256)
void proj_out_kernel(const bf16_t* __restrict__ Attn, const bf16_t* __restrict__ Wtp,
                     const float* __restrict__ bp, float* __restrict__ out) {
    int lane = threadIdx.x & 31;
    int wid  = threadIdx.x >> 5;
    int n0 = blockIdx.x * 16;
    int c0 = wid * 16;
    v8f acc = {};
#pragma unroll
    for (int k0 = 0; k0 < QKV_COLS; k0 += 32) {
        v16bf a = load_a_frag(Attn, QKV_COLS, n0, k0, lane);
        v16bf b = load_b_frag(Wtp, QKV_COLS, c0, k0, lane);
        acc = wmma_bf16(a, b, acc);
    }
    int half = lane >> 4, cl = lane & 15;
    float bias = bp[c0 + cl];
#pragma unroll
    for (int r = 0; r < 8; ++r)
        out[(size_t)(n0 + r + half * 8) * HID + c0 + cl] = acc[r] + bias;
}

// ---------------------------------------------------------------------------
// launcher
// ---------------------------------------------------------------------------
extern "C" void kernel_launch(void* const* d_in, const int* in_sizes, int n_in,
                              void* d_out, int out_size, void* d_ws, size_t ws_size,
                              hipStream_t stream) {
    const float*     H  = (const float*)d_in[0];
    const long long* ei = (const long long*)d_in[1];
    const float*     Wq = (const float*)d_in[2];
    const float*     Wk = (const float*)d_in[3];
    const float*     Wv = (const float*)d_in[4];
    const float*     Wp = (const float*)d_in[5];
    const float*     bp = (const float*)d_in[6];
    float*           out = (float*)d_out;

    // ---- workspace layout (256B aligned offsets), ~32 MB total
    char* ws = (char*)d_ws;
    unsigned int* mask = (unsigned int*)(ws);                        //  4.72 MB
    bf16_t* Qb    = (bf16_t*)(ws + 4718592);                         //  6.29 MB
    bf16_t* Kb    = (bf16_t*)(ws + 11010048);                        //  6.29 MB
    bf16_t* Vt    = (bf16_t*)(ws + 17301504);                        //  6.29 MB
    bf16_t* Attn  = (bf16_t*)(ws + 23592960);                        //  6.29 MB
    bf16_t* Hb    = (bf16_t*)(ws + 29884416);                        //  1.57 MB
    bf16_t* WtAll = (bf16_t*)(ws + 31457280);                        //  0.39 MB (3x512x128)
    bf16_t* Wtp   = (bf16_t*)(ws + 31850496);                        //  0.13 MB
    (void)ws_size; (void)n_in; (void)out_size;

    const int E = in_sizes[1] / 2;                       // edges incl. self loops
    const int MW = N_NODES * MASK_WPR;                   // mask words

    zero_u32_kernel<<<(MW + 255) / 256, 256, 0, stream>>>(mask, MW);
    edge_scatter_kernel<<<(E + 255) / 256, 256, 0, stream>>>(ei, E, mask);

    conv_f2bf_kernel<<<(N_NODES * HID + 255) / 256, 256, 0, stream>>>(H, Hb, N_NODES * HID);
    conv_transpose_kernel<<<(HID * QKV_COLS + 255) / 256, 256, 0, stream>>>(Wq, WtAll,                  HID, QKV_COLS);
    conv_transpose_kernel<<<(HID * QKV_COLS + 255) / 256, 256, 0, stream>>>(Wk, WtAll + 512 * HID,      HID, QKV_COLS);
    conv_transpose_kernel<<<(HID * QKV_COLS + 255) / 256, 256, 0, stream>>>(Wv, WtAll + 2 * 512 * HID,  HID, QKV_COLS);
    conv_transpose_kernel<<<(QKV_COLS * HID + 255) / 256, 256, 0, stream>>>(Wp, Wtp, QKV_COLS, HID);

    proj_qkv_kernel<<<dim3(N_NODES / 16, 12), 256, 0, stream>>>(Hb, WtAll, Qb, Kb, Vt);
    attn_kernel<<<N_NODES / 16, 128, 0, stream>>>(Qb, Kb, Vt, mask, Attn);
    proj_out_kernel<<<N_NODES / 16, 256, 0, stream>>>(Attn, Wtp, bp, out);
}
